// LFM2Attention_4818953306182
// MI455X (gfx1250) — matrix operand order repaired and checked by reading, not compiled, not run on hardware
//
#include <hip/hip_runtime.h>

// ---------------------------------------------------------------------------
// LFM2 attention for MI455X (gfx1250, wave32, WMMA bf16 16x16x32, f32 acc)
//   H=32, KVH=8, HD=64, D=2048, B=2, S=2048
// ---------------------------------------------------------------------------

#define Hh   32
#define KVH  8
#define HD   64
#define Dd   2048
#define Bb   2
#define Ss   2048
#define Mrows (Bb * Ss)              // 4096

typedef __attribute__((ext_vector_type(16))) __bf16 bf16x16;
typedef __attribute__((ext_vector_type(8)))  float  v8f;

__device__ __forceinline__ __bf16 f2bf(float x) {
    union { float f; unsigned u; } a; a.f = x;
    unsigned r = a.u + 0x7FFFu + ((a.u >> 16) & 1u);
    unsigned short h = (unsigned short)(r >> 16);
    union { unsigned short s; __bf16 b; } c; c.s = h;
    return c.b;
}

union FragU { bf16x16 v; uint4 q[2]; };

// A-fragment: lo = 8 contiguous bf16 (VGPRs 0-3), hi = 8 contiguous (VGPRs 4-7)
__device__ __forceinline__ bf16x16 load_frag2(const __bf16* lo, const __bf16* hi) {
    FragU f;
    f.q[0] = *(const uint4*)lo;
    f.q[1] = *(const uint4*)hi;
    return f.v;
}
// B-fragment: 16 contiguous bf16 along K (N-major storage)
__device__ __forceinline__ bf16x16 load_frag1(const __bf16* p) {
    return load_frag2(p, p + 8);
}

__device__ __forceinline__ v8f wmma_bf16(bf16x16 a, bf16x16 b, v8f c) {
    return __builtin_amdgcn_wmma_f32_16x16x32_bf16(false, a, false, b, (short)0, c,
                                                   false, false);
}

__device__ __forceinline__ v8f zero8() {
    v8f z = {0.f, 0.f, 0.f, 0.f, 0.f, 0.f, 0.f, 0.f};
    return z;
}

// --------------------------- fp32 -> bf16 convert ---------------------------
__global__ void k_cvt(const float* __restrict__ s, __bf16* __restrict__ d, int n) {
    int i = blockIdx.x * blockDim.x + threadIdx.x;
    int stride = gridDim.x * blockDim.x;
    for (; i < n; i += stride) d[i] = f2bf(s[i]);
}

// tiled transpose + convert: src [K][N] fp32 -> dst [N][K] bf16 (both coalesced)
__global__ __launch_bounds__(256) void k_cvtT(const float* __restrict__ s,
                                              __bf16* __restrict__ d,
                                              int K, int N) {
    __shared__ __bf16 tile[32][34];
    const int k0 = blockIdx.x * 32, n0 = blockIdx.y * 32;
    const int tx = threadIdx.x & 31, ty = threadIdx.x >> 5;   // 32 x 8
#pragma unroll
    for (int j = 0; j < 32; j += 8)
        tile[ty + j][tx] = f2bf(s[(size_t)(k0 + ty + j) * N + n0 + tx]);
    __syncthreads();
#pragma unroll
    for (int j = 0; j < 32; j += 8)
        d[(size_t)(n0 + ty + j) * K + k0 + tx] = tile[tx][ty + j];
}

// ------------------- fused QKV GEMM + RMSNorm + RoPE ------------------------
// grid (Mrows/128, 48): slot<32 -> Q head, 32..39 -> K head, 40..47 -> V head.
// One wave computes a 32-row x 64-col (one head) tile: 8 WMMA per 32-k step,
// 6 x b128-pair loads per 8 WMMA (A shared by 4 N-subtiles, B by 2 row-groups).
__global__ __launch_bounds__(128) void k_qkv(
    const __bf16* __restrict__ Xb,
    const __bf16* __restrict__ WqT, const __bf16* __restrict__ WkT,
    const __bf16* __restrict__ WvT,
    const float* __restrict__ cosT, const float* __restrict__ sinT,
    const float* __restrict__ qlnw, const float* __restrict__ klnw,
    __bf16* __restrict__ Qb, __bf16* __restrict__ Kb, __bf16* __restrict__ Vtb)
{
    const int wid = threadIdx.x >> 5;
    const int lane = threadIdx.x & 31;
    const int ln = lane & 15;
    const int g  = lane >> 4;
    const int m0 = blockIdx.x * 128 + wid * 32;    // first of 32 output rows
    const int ci = blockIdx.y;

    int type, hh;
    const __bf16* Bt;
    if (ci < 32)       { type = 0; hh = ci;      Bt = WqT + (size_t)hh * 64 * Dd; }
    else if (ci < 40)  { type = 1; hh = ci - 32; Bt = WkT + (size_t)hh * 64 * Dd; }
    else               { type = 2; hh = ci - 40; Bt = WvT + (size_t)hh * 64 * Dd; }

    const __bf16* arow[2];
#pragma unroll
    for (int rh = 0; rh < 2; ++rh) arow[rh] = Xb + (size_t)(m0 + 16 * rh + ln) * Dd;
    const __bf16* brow[4];
#pragma unroll
    for (int t = 0; t < 4; ++t) brow[t] = Bt + (size_t)(16 * t + ln) * Dd;

    v8f c[2][4];
#pragma unroll
    for (int rh = 0; rh < 2; ++rh)
#pragma unroll
        for (int t = 0; t < 4; ++t) c[rh][t] = zero8();

    for (int k0 = 0; k0 < Dd; k0 += 32) {
        __builtin_prefetch(arow[0] + k0 + 256, 0, 0);
        __builtin_prefetch(arow[1] + k0 + 256, 0, 0);
        __builtin_prefetch(brow[0] + k0 + 256, 0, 0);
        bf16x16 a0 = load_frag2(arow[0] + k0 + 8 * g, arow[0] + k0 + 16 + 8 * g);
        bf16x16 a1 = load_frag2(arow[1] + k0 + 8 * g, arow[1] + k0 + 16 + 8 * g);
#pragma unroll
        for (int t = 0; t < 4; ++t) {
            bf16x16 b = load_frag2(brow[t] + k0 + 8 * g, brow[t] + k0 + 16 + 8 * g);
            c[0][t] = wmma_bf16(a0, b, c[0][t]);
            c[1][t] = wmma_bf16(a1, b, c[1][t]);
        }
    }

    const int bidx = m0 >> 11;          // batch (S=2048 rows per batch)
    const int s0   = m0 & (Ss - 1);

    if (type == 2) {  // V: store transposed [B][KVH][HD][S], 16B packed stores
#pragma unroll
        for (int t = 0; t < 4; ++t) {
            int dcol = 16 * t + ln;
            __bf16* vp = Vtb + (((size_t)bidx * KVH + hh) * HD + dcol) * Ss;
#pragma unroll
            for (int rh = 0; rh < 2; ++rh) {
                union { __bf16 b[8]; uint4 u; } pk;
#pragma unroll
                for (int r = 0; r < 8; ++r) pk.b[r] = f2bf(c[rh][t][r]);
                *(uint4*)(vp + s0 + 16 * rh + 8 * g) = pk.u;
            }
        }
        return;
    }

    const float* lnw = (type == 0) ? qlnw : klnw;
    float w[4];
#pragma unroll
    for (int t = 0; t < 4; ++t) w[t] = lnw[16 * t + ln];
    const float QSCALE = 0.125f * 1.4426950408889634f;  // softmax scale * log2(e)

#pragma unroll
    for (int rh = 0; rh < 2; ++rh) {
        // per-row RMSNorm over the head's 64 columns (fp32, eps=1e-6)
        float inv[8];
#pragma unroll
        for (int r = 0; r < 8; ++r) {
            float ss = c[rh][0][r]*c[rh][0][r] + c[rh][1][r]*c[rh][1][r]
                     + c[rh][2][r]*c[rh][2][r] + c[rh][3][r]*c[rh][3][r];
            ss += __shfl_xor(ss, 8, 16);
            ss += __shfl_xor(ss, 4, 16);
            ss += __shfl_xor(ss, 2, 16);
            ss += __shfl_xor(ss, 1, 16);
            inv[r] = rsqrtf(ss * (1.0f / 64.0f) + 1e-6f);
        }
        float q[4][8];
#pragma unroll
        for (int t = 0; t < 4; ++t)
#pragma unroll
            for (int r = 0; r < 8; ++r) q[t][r] = c[rh][t][r] * inv[r] * w[t];

        // RoPE + store
#pragma unroll
        for (int t = 0; t < 4; ++t) {
            int dcol = 16 * t + ln;
#pragma unroll
            for (int r = 0; r < 8; ++r) {
                int srow = s0 + 16 * rh + r + 8 * g;
                float ce = cosT[srow * HD + dcol];
                float se = sinT[srow * HD + dcol];
                float rot = (t < 2) ? -q[t + 2][r] : q[t - 2][r];
                float val = q[t][r] * ce + rot * se;
                if (type == 0) {
                    Qb[(((size_t)bidx * Hh + hh) * Ss + srow) * HD + dcol] =
                        f2bf(val * QSCALE);
                } else {
                    Kb[(((size_t)bidx * KVH + hh) * Ss + srow) * HD + dcol] = f2bf(val);
                }
            }
        }
    }
}

// --------------------- causal flash attention (bf16 WMMA) -------------------
// grid (S/32, KVH, B); block = 4 waves = 4 query heads sharing ONE kv head and
// the SAME 32 query rows -> all waves issue identical K/V addresses (WGP$ hits)
// and identical trip counts.  Per 32-key step: 8 QK^T WMMA + 8 PV WMMA.
__global__ __launch_bounds__(128) void k_attn(
    const __bf16* __restrict__ Qb, const __bf16* __restrict__ Kb,
    const __bf16* __restrict__ Vtb, __bf16* __restrict__ AOb)
{
    __shared__ __bf16 pbuf[4][32 * 40];   // per-wave P staging (C-layout -> A-layout)

    const int wid = threadIdx.x >> 5;
    const int lane = threadIdx.x & 31;
    const int ln = lane & 15;
    const int g  = lane >> 4;
    const int qb = blockIdx.x, kvh = blockIdx.y, b = blockIdx.z;
    const int h = kvh * 4 + wid;                 // this wave's query head
    const int sq0 = qb * 32;

    bf16x16 aq[2][2];
#pragma unroll
    for (int rh = 0; rh < 2; ++rh) {
        const __bf16* qrow = Qb + (((size_t)b * Hh + h) * Ss + sq0 + 16 * rh + ln) * HD;
        aq[rh][0] = load_frag2(qrow +      8 * g, qrow + 16 + 8 * g);
        aq[rh][1] = load_frag2(qrow + 32 + 8 * g, qrow + 48 + 8 * g);
    }

    const __bf16* kbase = Kb  + ((size_t)b * KVH + kvh) * Ss * HD;
    const __bf16* vbase = Vtb + ((size_t)b * KVH + kvh) * HD * Ss;

    v8f o[2][4];
#pragma unroll
    for (int rh = 0; rh < 2; ++rh)
#pragma unroll
        for (int t = 0; t < 4; ++t) o[rh][t] = zero8();
    const float NEG = -__builtin_inff();
    float mrow[2][8], lrow[2][8];
#pragma unroll
    for (int rh = 0; rh < 2; ++rh)
#pragma unroll
        for (int r = 0; r < 8; ++r) { mrow[rh][r] = NEG; lrow[rh][r] = 0.f; }

    __bf16* pw = &pbuf[wid][0];
    const int nkb = qb + 1;                      // exact causal trip count

    for (int kb = 0; kb < nkb; ++kb) {
        const int kk0 = kb * 32;
        const __bf16* k0row = kbase + (size_t)(kk0 + ln) * HD;
        const __bf16* k1row = kbase + (size_t)(kk0 + 16 + ln) * HD;
        __builtin_prefetch(k0row + 2048, 0, 0);  // next key block
        __builtin_prefetch(k1row + 2048, 0, 0);

        bf16x16 bk00 = load_frag2(k0row +      8 * g, k0row + 16 + 8 * g);
        bf16x16 bk01 = load_frag2(k0row + 32 + 8 * g, k0row + 48 + 8 * g);
        bf16x16 bk10 = load_frag2(k1row +      8 * g, k1row + 16 + 8 * g);
        bf16x16 bk11 = load_frag2(k1row + 32 + 8 * g, k1row + 48 + 8 * g);

#pragma unroll
        for (int rh = 0; rh < 2; ++rh) {
            v8f s0 = wmma_bf16(aq[rh][0], bk00, zero8());
            s0     = wmma_bf16(aq[rh][1], bk01, s0);
            v8f s1 = wmma_bf16(aq[rh][0], bk10, zero8());
            s1     = wmma_bf16(aq[rh][1], bk11, s1);

#pragma unroll
            for (int r = 0; r < 8; ++r) {
                const int qi = sq0 + 16 * rh + r + 8 * g;
                float x0 = (kk0 + ln      <= qi) ? s0[r] : NEG;   // causal mask
                float x1 = (kk0 + 16 + ln <= qi) ? s1[r] : NEG;
                float mx = fmaxf(x0, x1);
                mx = fmaxf(mx, __shfl_xor(mx, 8, 16));
                mx = fmaxf(mx, __shfl_xor(mx, 4, 16));
                mx = fmaxf(mx, __shfl_xor(mx, 2, 16));
                mx = fmaxf(mx, __shfl_xor(mx, 1, 16));
                float mnew = fmaxf(mrow[rh][r], mx);
                float corr = exp2f(mrow[rh][r] - mnew);
                float p0 = exp2f(x0 - mnew);
                float p1 = exp2f(x1 - mnew);
                float ps = p0 + p1;
                ps += __shfl_xor(ps, 8, 16);
                ps += __shfl_xor(ps, 4, 16);
                ps += __shfl_xor(ps, 2, 16);
                ps += __shfl_xor(ps, 1, 16);
                lrow[rh][r] = lrow[rh][r] * corr + ps;
                mrow[rh][r] = mnew;
                o[rh][0][r] *= corr; o[rh][1][r] *= corr;
                o[rh][2][r] *= corr; o[rh][3][r] *= corr;
                const int mi = 16 * rh + r + 8 * g;
                pw[mi * 40 + ln]      = f2bf(p0);
                pw[mi * 40 + 16 + ln] = f2bf(p1);
            }
        }

        // Per-wave private LDS buffer + in-order DS pipeline (ISA 7.1): a
        // dscnt wait is sufficient, no workgroup barrier needed.
        asm volatile("s_wait_dscnt 0x0" ::: "memory");

#pragma unroll
        for (int td = 0; td < 4; ++td) {
            bf16x16 bv = load_frag2(vbase + (size_t)(16 * td + ln) * Ss + kk0 + 8 * g, vbase + (size_t)(16 * td + ln) * Ss + kk0 + 16 + 8 * g);
#pragma unroll
            for (int rh = 0; rh < 2; ++rh) {
                const __bf16* prow = pw + (16 * rh + ln) * 40;
                bf16x16 ap = load_frag2(prow + 8 * g, prow + 16 + 8 * g);
                o[rh][td] = wmma_bf16(ap, bv, o[rh][td]);
            }
        }
    }

    // normalize and store [B][S][D] bf16 (row-major in D: next GEMM's A operand)
#pragma unroll
    for (int rh = 0; rh < 2; ++rh)
#pragma unroll
        for (int r = 0; r < 8; ++r) {
            float invl = 1.0f / lrow[rh][r];
            int srow = sq0 + 16 * rh + r + 8 * g;
#pragma unroll
            for (int td = 0; td < 4; ++td) {
                int dcol = h * HD + 16 * td + ln;
                AOb[((size_t)b * Ss + srow) * Dd + dcol] = f2bf(o[rh][td][r] * invl);
            }
        }
}

// --------------------------- output projection ------------------------------
// One wave computes a 32-row x 64-col tile (same loop structure as k_qkv).
__global__ __launch_bounds__(128) void k_out(
    const __bf16* __restrict__ AOb, const __bf16* __restrict__ WoT,
    float* __restrict__ out)
{
    const int wid = threadIdx.x >> 5;
    const int lane = threadIdx.x & 31;
    const int ln = lane & 15;
    const int g  = lane >> 4;
    const int m0 = blockIdx.x * 128 + wid * 32;
    const int n0 = blockIdx.y * 64;

    const __bf16* arow[2];
#pragma unroll
    for (int rh = 0; rh < 2; ++rh) arow[rh] = AOb + (size_t)(m0 + 16 * rh + ln) * Dd;
    const __bf16* brow[4];
#pragma unroll
    for (int t = 0; t < 4; ++t) brow[t] = WoT + (size_t)(n0 + 16 * t + ln) * Dd;

    v8f c[2][4];
#pragma unroll
    for (int rh = 0; rh < 2; ++rh)
#pragma unroll
        for (int t = 0; t < 4; ++t) c[rh][t] = zero8();

    for (int k0 = 0; k0 < Dd; k0 += 32) {
        __builtin_prefetch(arow[0] + k0 + 256, 0, 0);
        __builtin_prefetch(arow[1] + k0 + 256, 0, 0);
        __builtin_prefetch(brow[0] + k0 + 256, 0, 0);
        bf16x16 a0 = load_frag2(arow[0] + k0 + 8 * g, arow[0] + k0 + 16 + 8 * g);
        bf16x16 a1 = load_frag2(arow[1] + k0 + 8 * g, arow[1] + k0 + 16 + 8 * g);
#pragma unroll
        for (int t = 0; t < 4; ++t) {
            bf16x16 b = load_frag2(brow[t] + k0 + 8 * g, brow[t] + k0 + 16 + 8 * g);
            c[0][t] = wmma_bf16(a0, b, c[0][t]);
            c[1][t] = wmma_bf16(a1, b, c[1][t]);
        }
    }

#pragma unroll
    for (int rh = 0; rh < 2; ++rh)
#pragma unroll
        for (int t = 0; t < 4; ++t) {
            int n = n0 + 16 * t + ln;
#pragma unroll
            for (int r = 0; r < 8; ++r)
                out[(size_t)(m0 + 16 * rh + r + 8 * g) * Dd + n] = c[rh][t][r];
        }
}

// ---------------------------------------------------------------------------
extern "C" void kernel_launch(void* const* d_in, const int* in_sizes, int n_in,
                              void* d_out, int out_size, void* d_ws, size_t ws_size,
                              hipStream_t stream) {
    (void)in_sizes; (void)n_in; (void)out_size; (void)ws_size;
    const float* X    = (const float*)d_in[0];
    const float* cosT = (const float*)d_in[1];
    const float* sinT = (const float*)d_in[2];
    const float* wq   = (const float*)d_in[3];
    const float* wk   = (const float*)d_in[4];
    const float* wv   = (const float*)d_in[5];
    const float* wo   = (const float*)d_in[6];
    const float* qlnw = (const float*)d_in[7];
    const float* klnw = (const float*)d_in[8];
    float* out = (float*)d_out;

    __bf16* ws  = (__bf16*)d_ws;
    __bf16* Xb  = ws;                       // 4096*2048        = 8388608
    __bf16* WqT = Xb  + 8388608;            // 2048*2048        = 4194304
    __bf16* WkT = WqT + 4194304;            // 512*2048         = 1048576
    __bf16* WvT = WkT + 1048576;            // 512*2048         = 1048576
    __bf16* WoT = WvT + 1048576;            // 2048*2048        = 4194304
    __bf16* Qb  = WoT + 4194304;            // 2*32*2048*64     = 8388608
    __bf16* Kb  = Qb  + 8388608;            // 2*8*2048*64      = 2097152
    __bf16* Vt  = Kb  + 2097152;            // 2*8*2048*64      = 2097152
    __bf16* AO  = Vt  + 2097152;            // 2*2048*2048      = 8388608

    k_cvt <<<2048, 256, 0, stream>>>(X,  Xb, 8388608);
    k_cvtT<<<dim3(64, 64), 256, 0, stream>>>(wq, WqT, 2048, 2048);
    k_cvtT<<<dim3(64, 16), 256, 0, stream>>>(wk, WkT, 2048, 512);
    k_cvtT<<<dim3(64, 16), 256, 0, stream>>>(wv, WvT, 2048, 512);
    k_cvtT<<<dim3(64, 64), 256, 0, stream>>>(wo, WoT, 2048, 2048);

    k_qkv <<<dim3(Mrows / 128, 48),     128, 0, stream>>>(Xb, WqT, WkT, WvT,
                                                          cosT, sinT, qlnw, klnw,
                                                          Qb, Kb, Vt);
    k_attn<<<dim3(Ss / 32, KVH, Bb),    128, 0, stream>>>(Qb, Kb, Vt, AO);
    k_out <<<dim3(Mrows / 128, Dd / 64), 128, 0, stream>>>(AO, WoT, out);
}
